// VectorQuantize_18202071400942
// MI455X (gfx1250) — compile-verified, matrix operand order checked by reading, not run on hardware
//
#include <hip/hip_runtime.h>
#include <hip/hip_bf16.h>
#include <stdint.h>

#define N_ROWS 32768   // 8*64*64
#define DIM    256
#define N_EMB  8192
#define MT     128     // rows per workgroup
#define NT     64      // codes per N-tile
#define THREADS 128    // 4 waves; each wave owns 32 rows x 64 cols
#define XS_STRIDE 264  // 256 + 8 pad (bf16 elems) -> 528B rows, conflict-free b128
#define ES_STRIDE 264

#define AS1 __attribute__((address_space(1)))
#define AS3 __attribute__((address_space(3)))

#if __has_builtin(__builtin_amdgcn_global_load_async_to_lds_b128)
#define HAVE_ASYNC_LDS 1
#else
#define HAVE_ASYNC_LDS 0
#endif

typedef int v4i __attribute__((vector_size(16)));   // matches builtin param: int __vector(4)
typedef __attribute__((ext_vector_type(16))) __bf16 bf16x16;
typedef __attribute__((ext_vector_type(8)))  float  f32x8;

__device__ __forceinline__ unsigned short f32_to_bf16_rne(float f) {
  union { float f; uint32_t u; } c; c.f = f;
  uint32_t u = c.u;
  uint32_t r = u + 0x7FFFu + ((u >> 16) & 1u);   // round-to-nearest-even
  return (unsigned short)(r >> 16);
}

// Load a 16x32 bf16 A/B fragment row for this lane from LDS.
// rowbase points at (matrix_row, kstep*32) in a row-major bf16 array.
// half = lane>=16: selects K chunks {8..15, 24..31} instead of {0..7, 16..23}.
__device__ __forceinline__ bf16x16 load_frag_lds(const unsigned short* rowbase, int half) {
  const uint4* p0 = reinterpret_cast<const uint4*>(rowbase + half * 8);
  const uint4* p1 = reinterpret_cast<const uint4*>(rowbase + half * 8 + 16);
  union { uint4 u[2]; bf16x16 v; } c;
  c.u[0] = *p0;
  c.u[1] = *p1;
  return c.v;
}

// Issue the copy of one NT x 512B E tile (pre-transposed bf16) into an LDS buffer.
// 2048 b128 chunks / THREADS = 16 per thread; async (fire-and-forget) when available.
__device__ __forceinline__ void stage_tile(const unsigned short* __restrict__ Et,
                                           unsigned short* dstbuf, int j0, int tid,
                                           bool prefetch_next) {
  for (int c = tid; c < NT * (DIM / 8); c += THREADS) {
    int j = c >> 5;            // row within tile
    int o = (c & 31) * 8;      // ushort offset within row (16B chunks)
    const unsigned short* src = Et + (size_t)(j0 + j) * DIM + o;
#if HAVE_ASYNC_LDS
    __builtin_amdgcn_global_load_async_to_lds_b128(
        (AS1 v4i*)src, (AS3 v4i*)(dstbuf + j * ES_STRIDE + o), 0, 0);
#else
    *(uint4*)(dstbuf + j * ES_STRIDE + o) = *(const uint4*)src;
#endif
    if (prefetch_next && ((c & 7) == 0))
      __builtin_prefetch(Et + (size_t)(j0 + NT + j) * DIM + o, 0, 1); // tile+2 -> L2
  }
}

__device__ __forceinline__ void wait_async_copies() {
#if HAVE_ASYNC_LDS
#if __has_builtin(__builtin_amdgcn_s_wait_asynccnt)
  __builtin_amdgcn_s_wait_asynccnt(0);
#else
  asm volatile("s_wait_asynccnt 0" ::: "memory");
#endif
#endif
}

// ---------------- Kernel 1: per-code squared norms ----------------
__global__ __launch_bounds__(256) void vq_norms(const float* __restrict__ embed,
                                                float* __restrict__ norms) {
  int j = blockIdx.x * 256 + threadIdx.x;
  float s = 0.f;
#pragma unroll 4
  for (int d = 0; d < DIM; ++d) {
    float e = embed[(size_t)d * N_EMB + j];
    s += e * e;
  }
  norms[j] = s;
}

// ---------------- Kernel 1b: one-time transpose + fp32->bf16 of the codebook ----
// Et[j][d] = bf16(E[d][j]); 64x64 LDS tiles, coalesced reads and writes.
__global__ __launch_bounds__(256) void vq_cvt_t(const float* __restrict__ E,
                                                unsigned short* __restrict__ Et) {
  __shared__ unsigned short tile[64][65];
  const int j0 = (blockIdx.x & 127) * 64;   // 8192/64 = 128 tiles along j
  const int d0 = (blockIdx.x >> 7) * 64;    // 256/64  = 4   tiles along d
  for (int idx = threadIdx.x; idx < 64 * 64; idx += 256) {
    int d = idx >> 6, j = idx & 63;
    tile[d][j] = f32_to_bf16_rne(E[(size_t)(d0 + d) * N_EMB + j0 + j]);
  }
  __syncthreads();
  for (int idx = threadIdx.x; idx < 64 * 64; idx += 256) {
    int j = idx >> 6, d = idx & 63;
    Et[(size_t)(j0 + j) * DIM + d0 + d] = tile[d][j];
  }
}

// ---------------- Kernel 2: fused GEMM (bf16 WMMA) + argmin, pipelined ----------
__global__ __launch_bounds__(THREADS) void vq_argmin(const float* __restrict__ X,
                                                     const unsigned short* __restrict__ Et,
                                                     const float* __restrict__ norms,
                                                     int*   __restrict__ ind_out,
                                                     float* __restrict__ indf_out) {
  extern __shared__ unsigned short lds[];
  unsigned short* Xs  = lds;                            // [MT][XS_STRIDE] bf16
  unsigned short* Es0 = lds + MT * XS_STRIDE;           // [NT][ES_STRIDE] bf16, buffer 0
  unsigned short* Es1 = Es0 + NT * ES_STRIDE;           // [NT][ES_STRIDE] bf16, buffer 1

  const int tid = threadIdx.x;
  const int n0  = blockIdx.x * MT;

  // Stage X tile fp32->bf16 (coalesced: consecutive tid -> consecutive d)
  for (int idx = tid; idx < MT * DIM; idx += THREADS) {
    int r = idx >> 8, d = idx & 255;
    Xs[r * XS_STRIDE + d] = f32_to_bf16_rne(X[(size_t)(n0 + r) * DIM + d]);
  }

  const int wave = tid >> 5;          // 0..3, owns rows [wave*32, wave*32+32)
  const int lane = tid & 31;
  const int half = lane >> 4;
  const int r16  = lane & 15;
  const unsigned short* arow0 = Xs + (wave * 32 +      r16) * XS_STRIDE;
  const unsigned short* arow1 = Xs + (wave * 32 + 16 + r16) * XS_STRIDE;

  float minval[2][8];
  int   minidx[2][8];
#pragma unroll
  for (int s = 0; s < 2; ++s)
#pragma unroll
    for (int v = 0; v < 8; ++v) { minval[s][v] = 3.4e38f; minidx[s][v] = 0; }

  const int NTILES = N_EMB / NT;

  // Prologue: kick off tile 0 into buffer 0.
  stage_tile(Et, Es0, 0, tid, true);

  for (int nt = 0; nt < NTILES; ++nt) {
    const int j0 = nt * NT;
    unsigned short* curb = (nt & 1) ? Es1 : Es0;
    unsigned short* nxtb = (nt & 1) ? Es0 : Es1;

    wait_async_copies();   // this wave's outstanding copies (into curb) done
    __syncthreads();       // all waves' copies done AND all waves done reading nxtb

    // Overlap: issue copy of tile nt+1 into the other buffer, then compute.
    if (nt + 1 < NTILES)
      stage_tile(Et, nxtb, j0 + NT, tid, nt + 2 < NTILES);

    f32x8 acc[2][4] = {};
#pragma unroll
    for (int ks = 0; ks < DIM / 32; ++ks) {
      bf16x16 a0 = load_frag_lds(arow0 + ks * 32, half);
      bf16x16 a1 = load_frag_lds(arow1 + ks * 32, half);
#pragma unroll
      for (int n = 0; n < 4; ++n) {
        bf16x16 b = load_frag_lds(curb + (n * 16 + r16) * ES_STRIDE + ks * 32, half);
        acc[0][n] = __builtin_amdgcn_wmma_f32_16x16x32_bf16(
            false, a0, false, b, (short)0, acc[0][n], false, false);
        acc[1][n] = __builtin_amdgcn_wmma_f32_16x16x32_bf16(
            false, a1, false, b, (short)0, acc[1][n], false, false);
      }
    }

    // score = ||e||^2 - 2*x.e  (||x||^2 is constant per row for argmin)
#pragma unroll
    for (int n = 0; n < 4; ++n) {
      int   j   = j0 + n * 16 + r16;      // C-layout: N = lane%16
      float nrm = norms[j];
#pragma unroll
      for (int s = 0; s < 2; ++s)
#pragma unroll
        for (int v = 0; v < 8; ++v) {     // C-layout: M = v + 8*half
          float sc = nrm - 2.0f * acc[s][n][v];
          if (sc < minval[s][v]) { minval[s][v] = sc; minidx[s][v] = j; }
        }
    }
  }

  // Cross-lane reduce over the 16 lanes sharing each M row (xor<=8 stays in half).
#pragma unroll
  for (int s = 0; s < 2; ++s)
#pragma unroll
    for (int v = 0; v < 8; ++v) {
      float bv = minval[s][v];
      int   bi = minidx[s][v];
#pragma unroll
      for (int off = 8; off > 0; off >>= 1) {
        float ov = __shfl_xor(bv, off, 32);
        int   oi = __shfl_xor(bi, off, 32);
        if (ov < bv || (ov == bv && oi < bi)) { bv = ov; bi = oi; }
      }
      if (r16 == 0) {
        int grow = n0 + wave * 32 + s * 16 + half * 8 + v;
        ind_out[grow]  = bi;
        indf_out[grow] = (float)bi;
      }
    }
}

// ---------------- Kernel 3: gather quantized rows + per-row SSE ----------------
__global__ __launch_bounds__(256) void vq_gather(const float* __restrict__ X,
                                                 const float* __restrict__ E,
                                                 const int*   __restrict__ ind,
                                                 float* __restrict__ qout,
                                                 float* __restrict__ partial) {
  __shared__ float red[256];
  const int row = blockIdx.x;
  const int t   = threadIdx.x;
  const int j   = ind[row];
  float q = E[(size_t)t * N_EMB + j];  // embed fits in L2; scattered but cached
  float x = X[(size_t)row * DIM + t];
  qout[(size_t)row * DIM + t] = q;
  float d = q - x;
  red[t] = d * d;
  __syncthreads();
#pragma unroll
  for (int s = 128; s > 0; s >>= 1) {
    if (t < s) red[t] += red[t + s];
    __syncthreads();
  }
  if (t == 0) partial[row] = red[0];
}

// ---------------- Kernel 4: deterministic loss reduction ----------------
__global__ __launch_bounds__(256) void vq_loss(const float* __restrict__ partial,
                                               float* __restrict__ loss) {
  __shared__ float red[256];
  const int t = threadIdx.x;
  float s = 0.f;
  for (int i = t; i < N_ROWS; i += 256) s += partial[i];
  red[t] = s;
  __syncthreads();
#pragma unroll
  for (int k = 128; k > 0; k >>= 1) {
    if (t < k) red[t] += red[t + k];
    __syncthreads();
  }
  if (t == 0) loss[0] = red[0] * (1.0f / ((float)N_ROWS * (float)DIM));
}

extern "C" void kernel_launch(void* const* d_in, const int* in_sizes, int n_in,
                              void* d_out, int out_size, void* d_ws, size_t ws_size,
                              hipStream_t stream) {
  const float* X = (const float*)d_in[0];   // [32768, 256] flattened
  const float* E = (const float*)d_in[1];   // [256, 8192]

  float* out   = (float*)d_out;
  float* qout  = out;                               // 8,388,608
  float* indf  = out + (size_t)N_ROWS * DIM;        // 32,768
  float* loss  = indf + N_ROWS;                     // 1

  float*          norms   = (float*)d_ws;                       // 8192 f32
  int*            indi    = (int*)(norms + N_EMB);              // 32768 i32
  float*          partial = (float*)(indi + N_ROWS);            // 32768 f32
  unsigned short* Et      = (unsigned short*)(partial + N_ROWS);// 8192*256 bf16 (16B-aligned)

  vq_norms<<<N_EMB / 256, 256, 0, stream>>>(E, norms);
  vq_cvt_t<<<(N_EMB / 64) * (DIM / 64), 256, 0, stream>>>(E, Et);

  size_t ldsBytes = (size_t)(MT * XS_STRIDE + 2 * NT * ES_STRIDE) * sizeof(unsigned short); // ~132KB
  vq_argmin<<<N_ROWS / MT, THREADS, ldsBytes, stream>>>(X, Et, norms, indi, indf);

  vq_gather<<<N_ROWS, 256, 0, stream>>>(X, E, indi, qout, partial);
  vq_loss<<<1, 256, 0, stream>>>(partial, loss);
}
